// ContextualBilateralLoss_40673340293860
// MI455X (gfx1250) — matrix operand order repaired and checked by reading, not compiled, not run on hardware
//
#include <hip/hip_runtime.h>
#include <math.h>

typedef __attribute__((ext_vector_type(16))) _Float16 v16h;
typedef __attribute__((ext_vector_type(8)))  _Float16 v8h;
typedef __attribute__((ext_vector_type(8)))  float    v8f;

constexpr int kC   = 512;
constexpr int kH   = 80;
constexpr int kW   = 80;
constexpr int kHW  = kH * kW;            // 6400
constexpr int kBM  = 64;                 // rows per block (A stripe in LDS)
constexpr int kNCH = 5;                  // j-chunks (parallelism, partial reductions)
constexpr int kJCH = kHW / kNCH;         // 1280 cols per chunk (5 iters of 256)
constexpr float kEPS = 1e-5f;
constexpr float kSPA = 2.0e5f / 6561.0f; // (2/bw) / ((H+1)^2 * eps_sp)

// ---------------------------------------------------------------------------
// Kernel 1: per-channel mean of y  ->  mu[c]
// ---------------------------------------------------------------------------
__global__ void channel_mean(const float* __restrict__ y, float* __restrict__ mu)
{
    const int c = blockIdx.x;
    const int t = threadIdx.x;
    float s = 0.0f;
    for (int i = t; i < kHW; i += 256) s += y[c * kHW + i];
    __shared__ float red[256];
    red[t] = s;
    __syncthreads();
    for (int off = 128; off > 0; off >>= 1) {
        if (t < off) red[t] += red[t + off];
        __syncthreads();
    }
    if (t == 0) mu[c] = red[0] * (1.0f / (float)kHW);
}

// ---------------------------------------------------------------------------
// Kernel 2: per-pixel L2 normalize (x - mu), (y - mu); pack f16 [pixel][ch]
// ---------------------------------------------------------------------------
__global__ void normalize_pack(const float* __restrict__ x, const float* __restrict__ y,
                               const float* __restrict__ mu,
                               _Float16* __restrict__ Xh, _Float16* __restrict__ Yh)
{
    const int t  = threadIdx.x;
    const int p  = t & 15;
    const int cg = t >> 4;
    const int i  = blockIdx.x * 16 + p;

    float sx = 0.0f, sy = 0.0f;
    for (int c = cg; c < kC; c += 16) {
        const float m  = mu[c];
        const float dx = x[c * kHW + i] - m;
        const float dy = y[c * kHW + i] - m;
        sx += dx * dx;
        sy += dy * dy;
    }
    __shared__ float rx[16][17], ry[16][17];
    __shared__ float ivx[16], ivy[16];
    rx[p][cg] = sx; ry[p][cg] = sy;
    __syncthreads();
    if (t < 16) {
        float ax = 0.0f, ay = 0.0f;
        for (int k = 0; k < 16; ++k) { ax += rx[t][k]; ay += ry[t][k]; }
        ivx[t] = 1.0f / fmaxf(sqrtf(ax), 1e-12f);   // torch F.normalize semantics
        ivy[t] = 1.0f / fmaxf(sqrtf(ay), 1e-12f);
    }
    __syncthreads();
    const float fx = ivx[p], fy = ivy[p];
    for (int c = cg; c < kC; c += 16) {
        const float m = mu[c];
        Xh[i * kC + c] = (_Float16)((x[c * kHW + i] - m) * fx);
        Yh[i * kC + c] = (_Float16)((y[c * kHW + i] - m) * fy);
    }
}

// ---------------------------------------------------------------------------
// WMMA sweep: 64-row blocks x 1280-col chunks (grid 100 x 5).
// Wave = 4 row-tiles (m) x 2 col-tiles (u): each A-frag LDS read feeds 2 WMMAs,
// each B-frag global read feeds 4 WMMAs. k-loop kept as a runtime loop so the
// A-frag ds_loads depend on k and cannot be LICM-hoisted out of the jb-loop
// (that hoisting is what caused scratch spills in the previous build).
// MODE 0: part[ch][i] = max_{j in chunk} cos[i,j]
// MODE 1: part[ch][i] = sum_{j in chunk} exp(2 - 2*(1-cos)/(dmin_i+eps))
// MODE 2: part[ch][i] = max_{j in chunk} (0.9*w/S_i + 0.1*w_sp/Ssp_i)
// ---------------------------------------------------------------------------
template <int MODE>
__global__ __launch_bounds__(256)
void cx_gemm(const _Float16* __restrict__ Xh, const _Float16* __restrict__ Yh,
             const float* __restrict__ dpart, const float* __restrict__ spart,
             float* __restrict__ outpart)
{
    __shared__ __align__(16) _Float16 Alds[kBM][520];  // 520-half pitch: bank spread
    __shared__ float red[8][kBM];
    __shared__ float cdL[kBM], csL[kBM], cspL[kBM];

    const int t    = threadIdx.x;
    const int wid  = t >> 5;
    const int lane = t & 31;
    const int hi   = lane >> 4;       // half-wave: K-split
    const int lo   = lane & 15;       // M (A) / N (B,C) lane index
    const int I0   = blockIdx.x * kBM;
    const int j0   = blockIdx.y * kJCH;

    // Stage A stripe (64 rows x 512 halves) into LDS: 128 halves per thread.
    {
        const int row = t >> 2;
        const int kst = (t & 3) * 128;
        const v8h* src = (const v8h*)(Xh + (I0 + row) * kC + kst);
        v8h* dst = (v8h*)&Alds[row][kst];
        #pragma unroll
        for (int q = 0; q < 16; ++q) dst[q] = src[q];
    }

    // Per-row epilogue constants into LDS (folds the partial-combine of the
    // previous pass: max of dmax partials / sum of S partials).
    if (MODE >= 1 && t < kBM) {
        const int i = I0 + t;
        float mx = -3.0e38f;
        for (int ch = 0; ch < kNCH; ++ch) mx = fmaxf(mx, dpart[ch * kHW + i]);
        cdL[t] = 1.0f / ((1.0f - mx) + kEPS);          // 1/(dmin + eps)
        if (MODE == 2) {
            float s = 0.0f;
            for (int ch = 0; ch < kNCH; ++ch) s += spart[ch * kHW + i];
            csL[t] = 1.0f / s;
            const float q  = __expf(-kSPA);
            const float E2 = __expf(2.0f);
            const int ri = i / kW, ci = i - ri * kW;
            const float Sr = 1.0f + q * (float)((ri > 0) + (ri < kH - 1));
            const float Sc = 1.0f + q * (float)((ci > 0) + (ci < kW - 1));
            cspL[t] = 1.0f / (E2 * Sr * Sc);           // 1/Ssp_i (exact in f32)
        }
    }
    __syncthreads();

    float run[4][8];
    #pragma unroll
    for (int m = 0; m < 4; ++m)
        #pragma unroll
        for (int r = 0; r < 8; ++r) run[m][r] = (MODE == 1) ? 0.0f : -3.0e38f;

    const int rowA = lo;
    #pragma clang loop unroll(disable)
    for (int jb = j0 + (wid << 5); jb < j0 + kJCH; jb += 256) {
        v8f acc[2][4];
        #pragma unroll
        for (int u = 0; u < 2; ++u)
            #pragma unroll
            for (int m = 0; m < 4; ++m) acc[u][m] = (v8f){};

        const _Float16* By1 = Yh + (jb + lo) * kC + (hi << 4);
        const _Float16* By2 = By1 + 16 * kC;

        #pragma unroll 2
        for (int k = 0; k < 16; ++k) {
            const v16h B1 = *(const v16h*)(By1 + (k << 5));
            const v16h B2 = *(const v16h*)(By2 + (k << 5));
            const int kA = (k << 5) + (hi << 4);
            #pragma unroll
            for (int m = 0; m < 4; ++m) {
                union { v16h v; v8h h[2]; } A;
                A.h[0] = *(const v8h*)&Alds[(m << 4) + rowA][kA];
                A.h[1] = *(const v8h*)&Alds[(m << 4) + rowA][kA + 8];
                acc[0][m] = __builtin_amdgcn_wmma_f32_16x16x32_f16(
                                false, A.v, false, B1, (short)0, acc[0][m], false, false);
                acc[1][m] = __builtin_amdgcn_wmma_f32_16x16x32_f16(
                                false, A.v, false, B2, (short)0, acc[1][m], false, false);
            }
        }

        #pragma unroll
        for (int u = 0; u < 2; ++u) {
            const int j = jb + (u << 4) + lo;
            int jr = 0, jc = 0;
            if (MODE == 2) { jr = j / kW; jc = j - jr * kW; }
            #pragma unroll
            for (int m = 0; m < 4; ++m) {
                #pragma unroll
                for (int r = 0; r < 8; ++r) {
                    const int row = (m << 4) + r + (hi << 3);
                    const float cosv = acc[u][m][r];
                    float val;
                    if (MODE == 0) {
                        val = cosv;
                    } else {
                        const float w = __expf(2.0f - 2.0f * (1.0f - cosv) * cdL[row]);
                        if (MODE == 1) {
                            val = w;
                        } else {
                            const int i  = I0 + row;
                            const int ri = i / kW, ci = i - ri * kW;
                            const int dr = ri - jr, dc = ci - jc;
                            const float wsp = __expf(2.0f - kSPA * (float)(dr * dr + dc * dc));
                            val = 0.9f * w * csL[row] + 0.1f * wsp * cspL[row];
                        }
                    }
                    if (MODE == 1) {
                        for (int s = 1; s < 16; s <<= 1) val += __shfl_xor(val, s, 32);
                        run[m][r] += val;
                    } else {
                        for (int s = 1; s < 16; s <<= 1) val = fmaxf(val, __shfl_xor(val, s, 32));
                        run[m][r] = fmaxf(run[m][r], val);
                    }
                }
            }
        }
    }

    if (lo == 0)
        #pragma unroll
        for (int m = 0; m < 4; ++m)
            #pragma unroll
            for (int r = 0; r < 8; ++r) red[wid][(m << 4) + r + (hi << 3)] = run[m][r];
    __syncthreads();
    if (t < kBM) {
        float g = red[0][t];
        for (int w = 1; w < 8; ++w)
            g = (MODE == 1) ? (g + red[w][t]) : fmaxf(g, red[w][t]);
        outpart[blockIdx.y * kHW + I0 + t] = g;
    }
}

// ---------------------------------------------------------------------------
// Final: kmax[i] = max over chunks; cx = mean_i kmax;  out = -log(cx + eps)
// ---------------------------------------------------------------------------
__global__ void finalize(const float* __restrict__ kpart, float* __restrict__ out)
{
    const int t = threadIdx.x;
    float s = 0.0f;
    for (int i = t; i < kHW; i += 256) {
        float m = -3.0e38f;
        for (int ch = 0; ch < kNCH; ++ch) m = fmaxf(m, kpart[ch * kHW + i]);
        s += m;
    }
    __shared__ float red[256];
    red[t] = s;
    __syncthreads();
    for (int off = 128; off > 0; off >>= 1) {
        if (t < off) red[t] += red[t + off];
        __syncthreads();
    }
    if (t == 0) out[0] = -logf(red[0] * (1.0f / (float)kHW) + kEPS);
}

// ---------------------------------------------------------------------------
extern "C" void kernel_launch(void* const* d_in, const int* in_sizes, int n_in,
                              void* d_out, int out_size, void* d_ws, size_t ws_size,
                              hipStream_t stream)
{
    (void)in_sizes; (void)n_in; (void)out_size; (void)ws_size;
    const float* x = (const float*)d_in[0];
    const float* y = (const float*)d_in[1];

    char* ws = (char*)d_ws;
    constexpr size_t kFeatBytes = (size_t)kHW * kC * 2;       // 6,553,600
    constexpr size_t kPartBytes = (size_t)kNCH * kHW * 4;     // 128,000
    float*    mu    = (float*)ws;                             // 512 f
    _Float16* Xh    = (_Float16*)(ws + 2048);
    _Float16* Yh    = (_Float16*)(ws + 2048 + kFeatBytes);
    float*    dmaxp = (float*)(ws + 2048 + 2 * kFeatBytes);
    float*    ssump = (float*)(ws + 2048 + 2 * kFeatBytes + kPartBytes);
    float*    kmaxp = (float*)(ws + 2048 + 2 * kFeatBytes + 2 * kPartBytes);
    float*    out   = (float*)d_out;

    const dim3 gGemm(kHW / kBM, kNCH);
    channel_mean  <<<kC,       256, 0, stream>>>(y, mu);
    normalize_pack<<<kHW / 16, 256, 0, stream>>>(x, y, mu, Xh, Yh);
    cx_gemm<0>    <<<gGemm,    256, 0, stream>>>(Xh, Yh, nullptr, nullptr, dmaxp);
    cx_gemm<1>    <<<gGemm,    256, 0, stream>>>(Xh, Yh, dmaxp, nullptr, ssump);
    cx_gemm<2>    <<<gGemm,    256, 0, stream>>>(Xh, Yh, dmaxp, ssump, kmaxp);
    finalize      <<<1,        256, 0, stream>>>(kmaxp, out);
}
